// Net52_15788299780227
// MI455X (gfx1250) — compile-verified, matrix operand order checked by reading, not run on hardware
//
#include <hip/hip_runtime.h>

#define DF 128

typedef __attribute__((ext_vector_type(16))) _Float16 v16h;
typedef __attribute__((ext_vector_type(8)))  float    v8f;

// ---------------- helpers: order-preserving float <-> uint for atomic max ----
__device__ __forceinline__ unsigned ordF(float f) {
  unsigned u = __float_as_uint(f);
  return (u & 0x80000000u) ? ~u : (u | 0x80000000u);
}
__device__ __forceinline__ float unordF(unsigned o) {
  return __uint_as_float((o & 0x80000000u) ? (o & 0x7fffffffu) : ~o);
}

// ---------------- trivial utility kernels ------------------------------------
__global__ void k_zero_f(float* __restrict__ p, int n) {
  int i = blockIdx.x * blockDim.x + threadIdx.x;
  if (i < n) p[i] = 0.0f;
}
__global__ void k_zero_u(unsigned* __restrict__ p, int n) {
  int i = blockIdx.x * blockDim.x + threadIdx.x;
  if (i < n) p[i] = 0u;
}

// out[i,:] = emb[ids[i],:]
__global__ void k_gather_rows(const float* __restrict__ emb, const int* __restrict__ ids,
                              float* __restrict__ out, int n) {
  long t = (long)blockIdx.x * blockDim.x + threadIdx.x;
  if (t >= (long)n * DF) return;
  int r = (int)(t >> 7), f = (int)(t & (DF - 1));
  out[t] = emb[(long)ids[r] * DF + f];
}

// ---------------- WMMA GEMM: C[M,128] = act(A[M,K] @ W[K,128] + bias) --------
// One wave -> one 16x16 tile; 8 waves/block cover all 8 column tiles of a
// 16-row stripe. W is staged into LDS *pre-swizzled into the WMMA B-fragment
// layout* ([ktile][ntile][lane][16 halves]) so each wave reads its fragment as
// one contiguous 32B v16h load (2x ds_load_b128) instead of 16 strided u16
// loads. Two accumulators break the WMMA->WMMA RAW chain (no v_nop slots).
template <int K>
__global__ __launch_bounds__(256) void k_gemm_wmma(
    const float* __restrict__ A, const float* __restrict__ W,
    const float* __restrict__ bias, float* __restrict__ C, int M, int doRelu) {
  extern __shared__ _Float16 Wl[];  // K*DF halves, swizzled

  // Stage + swizzle W:  idx = ((kt*8 + nt)*32 + lane)*16 + j
  for (int idx = threadIdx.x; idx < K * DF; idx += 256) {
    const int j = idx & 15;
    const int ln = (idx >> 4) & 31;
    const int tile = idx >> 9;       // kt*8 + nt
    const int nt = tile & 7;
    const int kt = tile >> 3;
    const int lmS = ln & 15, hiS = ln >> 4;
    const int colS = nt * 16 + lmS;
    const int kS = kt * 32 + ((j < 8) ? (j + 8 * hiS) : (8 + j + 8 * hiS));
    Wl[idx] = (_Float16)W[kS * DF + colS];
  }
  __syncthreads();

  const int wave = threadIdx.x >> 5;
  const int lane = threadIdx.x & 31;
  const int lm = lane & 15, hi = lane >> 4;
  const int m0 = blockIdx.x * 16;
  const int n0 = wave * 16;
  const int row = m0 + lm;
  const int arow = (row < M) ? row : (M - 1);  // clamp (stores are masked)
  const int col = n0 + lm;

  const float* __restrict__ Ar = A + (long)arow * K;
  // per-wave base of swizzled B fragments: + ktile*8*512 per k-step
  const _Float16* __restrict__ Bb = Wl + (wave * 32 + lane) * 16;

  v8f acc0 = {}, acc1 = {};

#pragma unroll
  for (int kt = 0; kt < K; kt += 32) {
    const int kb0 = kt + 8 * hi;        // lane's first contiguous 8-float run
    const int kb1 = kt + 16 + 8 * hi;   // lane's second run
    float4 p0 = *(const float4*)(Ar + kb0);
    float4 p1 = *(const float4*)(Ar + kb0 + 4);
    float4 p2 = *(const float4*)(Ar + kb1);
    float4 p3 = *(const float4*)(Ar + kb1 + 4);
    v16h a;
    a[0]  = (_Float16)p0.x; a[1]  = (_Float16)p0.y; a[2]  = (_Float16)p0.z; a[3]  = (_Float16)p0.w;
    a[4]  = (_Float16)p1.x; a[5]  = (_Float16)p1.y; a[6]  = (_Float16)p1.z; a[7]  = (_Float16)p1.w;
    a[8]  = (_Float16)p2.x; a[9]  = (_Float16)p2.y; a[10] = (_Float16)p2.z; a[11] = (_Float16)p2.w;
    a[12] = (_Float16)p3.x; a[13] = (_Float16)p3.y; a[14] = (_Float16)p3.z; a[15] = (_Float16)p3.w;

    v16h b = *(const v16h*)(Bb + (kt >> 5) * (8 * 512));  // contiguous 32B

    if ((kt >> 5) & 1)
      acc1 = __builtin_amdgcn_wmma_f32_16x16x32_f16(false, a, false, b, (short)0,
                                                    acc1, false, false);
    else
      acc0 = __builtin_amdgcn_wmma_f32_16x16x32_f16(false, a, false, b, (short)0,
                                                    acc0, false, false);
  }
  v8f acc = acc0 + acc1;

#pragma unroll
  for (int r = 0; r < 8; ++r) {
    int m = m0 + r + 8 * hi;
    if (m < M) {
      float v = acc[r];
      if (bias) v += bias[col];
      if (doRelu) v = v > 0.0f ? v : 0.0f;
      C[(long)m * DF + col] = v;
    }
  }
}

// hs[i] = dot(h[i,:], as), hd[i] = dot(h[i,:], ad)   (one block of 128 per row)
__global__ __launch_bounds__(128) void k_dots(
    const float* __restrict__ h, const float* __restrict__ as_,
    const float* __restrict__ ad_, float* __restrict__ hs, float* __restrict__ hd,
    int n) {
  __shared__ float r1[128], r2[128];
  int row = blockIdx.x, t = threadIdx.x;
  float v = h[(long)row * DF + t];
  r1[t] = v * as_[t];
  r2[t] = v * ad_[t];
  __syncthreads();
  for (int s = 64; s > 0; s >>= 1) {
    if (t < s) { r1[t] += r1[t + s]; r2[t] += r2[t + s]; }
    __syncthreads();
  }
  if (t == 0) { hs[row] = r1[0]; hd[row] = r2[0]; }
}

// pass 1: segment max of leaky_relu(hs[s]+hd[d]) into mOrd[d]
__global__ void k_edge_max(const int* __restrict__ src, const int* __restrict__ dst,
                           const float* __restrict__ hs, const float* __restrict__ hd,
                           unsigned* __restrict__ mOrd, int E, int n) {
  int i = blockIdx.x * blockDim.x + threadIdx.x;
  if (i >= E + n) return;
  int s = (i < E) ? src[i] : (i - E);
  int d = (i < E) ? dst[i] : (i - E);
  float e = hs[s] + hd[d];
  e = (e > 0.0f) ? e : 0.2f * e;
  atomicMax(&mOrd[d], ordF(e));
}

// pass 2: ex = exp(e - m[d]); z[d] += ex
__global__ void k_edge_expsum(const int* __restrict__ src, const int* __restrict__ dst,
                              const float* __restrict__ hs, const float* __restrict__ hd,
                              const unsigned* __restrict__ mOrd, float* __restrict__ ex,
                              float* __restrict__ z, int E, int n) {
  int i = blockIdx.x * blockDim.x + threadIdx.x;
  if (i >= E + n) return;
  int s = (i < E) ? src[i] : (i - E);
  int d = (i < E) ? dst[i] : (i - E);
  float e = hs[s] + hd[d];
  e = (e > 0.0f) ? e : 0.2f * e;
  float v = __expf(e - unordF(mOrd[d]));
  ex[i] = v;
  atomicAdd(&z[d], v);
}

// pass 3: out[d,:] += (ex/z[d]) * h[s,:]   (one thread per edge-feature)
__global__ void k_edge_agg(const int* __restrict__ src, const int* __restrict__ dst,
                           const float* __restrict__ ex, const float* __restrict__ z,
                           const float* __restrict__ h, float* __restrict__ out,
                           int E, int n) {
  long t = (long)blockIdx.x * blockDim.x + threadIdx.x;
  if (t >= (long)(E + n) * DF) return;
  int i = (int)(t >> 7), f = (int)(t & (DF - 1));
  int s = (i < E) ? src[i] : (i - E);
  int d = (i < E) ? dst[i] : (i - E);
  float w = ex[i] / z[d];
  atomicAdd(&out[(long)d * DF + f], w * h[(long)s * DF + f]);
}

// x = relu(x + b); score = dot(x,p)/||p||; x *= tanh(score)    (in place)
__global__ __launch_bounds__(128) void k_bias_relu_gate(
    float* __restrict__ x, const float* __restrict__ b, const float* __restrict__ p,
    int n) {
  __shared__ float red[128];
  __shared__ float pinv;
  int row = blockIdx.x, t = threadIdx.x;
  float pv = p[t];
  red[t] = pv * pv;
  __syncthreads();
  for (int s = 64; s > 0; s >>= 1) { if (t < s) red[t] += red[t + s]; __syncthreads(); }
  if (t == 0) pinv = rsqrtf(red[0] > 0.0f ? red[0] : 1e-30f);
  __syncthreads();
  float v = x[(long)row * DF + t] + b[t];
  v = v > 0.0f ? v : 0.0f;
  red[t] = v * pv;
  __syncthreads();
  for (int s = 64; s > 0; s >>= 1) { if (t < s) red[t] += red[t + s]; __syncthreads(); }
  float g = tanhf(red[0] * pinv);
  x[(long)row * DF + t] = v * g;
}

// segment max/sum/count per graph
__global__ void k_pool(const float* __restrict__ x, const int* __restrict__ batch,
                       unsigned* __restrict__ mx, float* __restrict__ sm,
                       float* __restrict__ cnt, int n) {
  long t = (long)blockIdx.x * blockDim.x + threadIdx.x;
  if (t >= (long)n * DF) return;
  int r = (int)(t >> 7), f = (int)(t & (DF - 1));
  int g = batch[r];
  float v = x[t];
  atomicMax(&mx[(long)g * DF + f], ordF(v));
  atomicAdd(&sm[(long)g * DF + f], v);
  if (f == 0) atomicAdd(&cnt[g], 1.0f);
}

// xsum[g, 0:128]   += max ;  xsum[g, 128:256] += sum/max(cnt,1)
__global__ void k_pool_fin(const unsigned* __restrict__ mx, const float* __restrict__ sm,
                           const float* __restrict__ cnt, float* __restrict__ xsum,
                           int G) {
  int t = blockIdx.x * blockDim.x + threadIdx.x;
  if (t >= G * DF) return;
  int g = t >> 7, f = t & (DF - 1);
  float c = cnt[g];
  float mxv = (c > 0.0f) ? unordF(mx[t]) : 0.0f;
  xsum[(long)g * (2 * DF) + f] += mxv;
  xsum[(long)g * (2 * DF) + DF + f] += sm[t] / (c > 1.0f ? c : 1.0f);
}

// xb = concat(xDP[0:nDP], emb[x_ids[skip + i]])
__global__ void k_build_xb(const float* __restrict__ xDP, const float* __restrict__ emb,
                           const int* __restrict__ x_ids, int skip,
                           float* __restrict__ xb, int nDP, int nTot) {
  long t = (long)blockIdx.x * blockDim.x + threadIdx.x;
  if (t >= (long)nTot * DF) return;
  int r = (int)(t >> 7), f = (int)(t & (DF - 1));
  xb[t] = (r < nDP) ? xDP[t] : emb[(long)x_ids[skip + (r - nDP)] * DF + f];
}

__global__ void k_deg(const int* __restrict__ src, const int* __restrict__ dst,
                      float* __restrict__ deg, int E, int n) {
  int i = blockIdx.x * blockDim.x + threadIdx.x;
  if (i >= E + n) return;
  (void)src;
  int d = (i < E) ? dst[i] : (i - E);
  atomicAdd(&deg[d], 1.0f);
}
__global__ void k_rsqrt(float* __restrict__ p, int n) {
  int i = blockIdx.x * blockDim.x + threadIdx.x;
  if (i < n) p[i] = rsqrtf(p[i] > 0.0f ? p[i] : 1.0f);
}

// out[d,:] += dis[s]*dis[d]*h[s,:]
__global__ void k_gcn_agg(const int* __restrict__ src, const int* __restrict__ dst,
                          const float* __restrict__ dis, const float* __restrict__ h,
                          float* __restrict__ out, int E, int n) {
  long t = (long)blockIdx.x * blockDim.x + threadIdx.x;
  if (t >= (long)(E + n) * DF) return;
  int i = (int)(t >> 7), f = (int)(t & (DF - 1));
  int s = (i < E) ? src[i] : (i - E);
  int d = (i < E) ? dst[i] : (i - E);
  atomicAdd(&out[(long)d * DF + f], dis[s] * dis[d] * h[(long)s * DF + f]);
}

__global__ void k_bias_relu(float* __restrict__ x, const float* __restrict__ b, int n) {
  long t = (long)blockIdx.x * blockDim.x + threadIdx.x;
  if (t >= (long)n * DF) return;
  int f = (int)(t & (DF - 1));
  float v = x[t] + b[f];
  x[t] = v > 0.0f ? v : 0.0f;
}

// out = concat(xb[drugNodes], xb[seNodes], xb)
__global__ void k_out(const float* __restrict__ xb, const int* __restrict__ dn,
                      const int* __restrict__ sn, float* __restrict__ out,
                      int nd, int ns, int nb) {
  long t = (long)blockIdx.x * blockDim.x + threadIdx.x;
  if (t >= (long)(nd + ns + nb) * DF) return;
  int r = (int)(t >> 7), f = (int)(t & (DF - 1));
  int srow = (r < nd) ? dn[r] : (r < nd + ns) ? sn[r - nd] : (r - nd - ns);
  out[t] = xb[(long)srow * DF + f];
}

// -----------------------------------------------------------------------------
extern "C" void kernel_launch(void* const* d_in, const int* in_sizes, int n_in,
                              void* d_out, int out_size, void* d_ws, size_t ws_size,
                              hipStream_t stream) {
  (void)n_in; (void)out_size; (void)ws_size;
  const int*   x_ids  = (const int*)d_in[0];
  const int*   drugE  = (const int*)d_in[1];
  const int*   drugN  = (const int*)d_in[3];
  const int*   seN    = (const int*)d_in[4];
  const int*   atom_x = (const int*)d_in[7];
  const int*   atomE  = (const int*)d_in[8];
  const int*   abatch = (const int*)d_in[9];
  const float* emb    = (const float*)d_in[11];
  const float* gatW[3]  = {(const float*)d_in[12], (const float*)d_in[17], (const float*)d_in[22]};
  const float* attS[3]  = {(const float*)d_in[13], (const float*)d_in[18], (const float*)d_in[23]};
  const float* attD[3]  = {(const float*)d_in[14], (const float*)d_in[19], (const float*)d_in[24]};
  const float* gatB[3]  = {(const float*)d_in[15], (const float*)d_in[20], (const float*)d_in[25]};
  const float* poolP[3] = {(const float*)d_in[16], (const float*)d_in[21], (const float*)d_in[26]};
  const float* lin1W = (const float*)d_in[27];
  const float* lin1b = (const float*)d_in[28];
  const float* gcnW[2] = {(const float*)d_in[29], (const float*)d_in[31]};
  const float* gcnB[2] = {(const float*)d_in[30], (const float*)d_in[32]};

  const int NA = in_sizes[7];        // 200000 atoms
  const int EA = in_sizes[8] / 2;    // 800000 atom edges
  const int NB = in_sizes[0];        // 7000 big-graph nodes
  const int EB = in_sizes[1] / 2;    // 200000 big-graph edges
  const int GG = 2000;               // graphs (reference constant G)
  const int skip = in_sizes[3] + in_sizes[5];  // nDrug + nProtein = 2000
  const int nd = in_sizes[3], ns = in_sizes[4];

  // ---- workspace carve (floats; all offsets multiples of 4 => 16B aligned) --
  float* w = (float*)d_ws;
  float* X = w;                    w += (long)NA * DF;   // node features (in/out)
  float* H = w;                    w += (long)NA * DF;   // x @ W
  float* hs = w;                   w += NA;
  float* hd = w;                   w += NA;
  unsigned* mOrd = (unsigned*)w;   w += NA;
  float* z = w;                    w += NA;
  float* ex = w;                   w += (long)(EA + NA);
  unsigned* mx = (unsigned*)w;     w += (long)GG * DF;
  float* sm = w;                   w += (long)GG * DF;
  float* cnt = w;                  w += GG;
  float* xsum = w;                 w += (long)GG * 2 * DF;
  float* xDP = w;                  w += (long)GG * DF;
  float* xb = w;                   w += (long)NB * DF;
  float* hb = w;                   w += (long)NB * DF;
  float* dis = w;                  w += NB;

  auto cdiv = [](long a, long b) { return (int)((a + b - 1) / b); };
  const int B = 256;
  const int* asrc = atomE;
  const int* adst = atomE + EA;
  const size_t sh128 = (size_t)128 * DF * sizeof(_Float16);  // 32 KB
  const size_t sh256 = (size_t)256 * DF * sizeof(_Float16);  // 64 KB

  // xa = emb[atom_x]
  k_gather_rows<<<cdiv((long)NA * DF, B), B, 0, stream>>>(emb, atom_x, X, NA);
  k_zero_f<<<cdiv(GG * 2 * DF, B), B, 0, stream>>>(xsum, GG * 2 * DF);

  // ---- 3 GAT layers ---------------------------------------------------------
  for (int l = 0; l < 3; ++l) {
    k_gemm_wmma<128><<<cdiv(NA, 16), 256, sh128, stream>>>(X, gatW[l], nullptr, H, NA, 0);
    k_dots<<<NA, 128, 0, stream>>>(H, attS[l], attD[l], hs, hd, NA);
    k_zero_u<<<cdiv(NA, B), B, 0, stream>>>(mOrd, NA);
    k_zero_f<<<cdiv(NA, B), B, 0, stream>>>(z, NA);
    k_edge_max<<<cdiv(EA + NA, B), B, 0, stream>>>(asrc, adst, hs, hd, mOrd, EA, NA);
    k_edge_expsum<<<cdiv(EA + NA, B), B, 0, stream>>>(asrc, adst, hs, hd, mOrd, ex, z, EA, NA);
    k_zero_f<<<cdiv((long)NA * DF, B), B, 0, stream>>>(X, NA * DF);   // X dead after GEMM
    k_edge_agg<<<cdiv((long)(EA + NA) * DF, B), B, 0, stream>>>(asrc, adst, ex, z, H, X, EA, NA);
    k_bias_relu_gate<<<NA, 128, 0, stream>>>(X, gatB[l], poolP[l], NA);
    k_zero_u<<<cdiv(GG * DF, B), B, 0, stream>>>(mx, GG * DF);
    k_zero_f<<<cdiv(GG * DF, B), B, 0, stream>>>(sm, GG * DF);
    k_zero_f<<<cdiv(GG, B), B, 0, stream>>>(cnt, GG);
    k_pool<<<cdiv((long)NA * DF, B), B, 0, stream>>>(X, abatch, mx, sm, cnt, NA);
    k_pool_fin<<<cdiv(GG * DF, B), B, 0, stream>>>(mx, sm, cnt, xsum, GG);
  }

  // xDP = relu(xsum @ lin1_W + lin1_b)   (K = 256)
  k_gemm_wmma<256><<<cdiv(GG, 16), 256, sh256, stream>>>(xsum, lin1W, lin1b, xDP, GG, 1);

  // xb = concat(xDP, emb[x_ids[2000:]])
  k_build_xb<<<cdiv((long)NB * DF, B), B, 0, stream>>>(xDP, emb, x_ids, skip, xb, GG, NB);

  // GCN degree normalization (shared by both layers)
  const int* bsrc = drugE;
  const int* bdst = drugE + EB;
  k_zero_f<<<cdiv(NB, B), B, 0, stream>>>(dis, NB);
  k_deg<<<cdiv(EB + NB, B), B, 0, stream>>>(bsrc, bdst, dis, EB, NB);
  k_rsqrt<<<cdiv(NB, B), B, 0, stream>>>(dis, NB);

  // ---- 2 GCN layers (xb updated in place via hb temp) -----------------------
  for (int l = 0; l < 2; ++l) {
    k_gemm_wmma<128><<<cdiv(NB, 16), 256, sh128, stream>>>(xb, gcnW[l], nullptr, hb, NB, 0);
    k_zero_f<<<cdiv((long)NB * DF, B), B, 0, stream>>>(xb, NB * DF);
    k_gcn_agg<<<cdiv((long)(EB + NB) * DF, B), B, 0, stream>>>(bsrc, bdst, dis, hb, xb, EB, NB);
    k_bias_relu<<<cdiv((long)NB * DF, B), B, 0, stream>>>(xb, gcnB[l], NB);
  }

  // outputs
  k_out<<<cdiv((long)(nd + ns + NB) * DF, B), B, 0, stream>>>(xb, drugN, seN,
                                                             (float*)d_out, nd, ns, NB);
}
